// Drug_PNAConv_29137058136191
// MI455X (gfx1250) — matrix operand order repaired and checked by reading, not compiled, not run on hardware
//
#include <hip/hip_runtime.h>
#include <math.h>

#define NN 40000
#define NE 320000
#define HD 128
#define NT 4
#define FT 32

typedef __attribute__((ext_vector_type(16))) __bf16 v16bf;
typedef __attribute__((ext_vector_type(8)))  float  v8f;
typedef __attribute__((ext_vector_type(8)))  unsigned int v8u;

// ---------- helpers ----------
__device__ __forceinline__ unsigned short bfc(float f) {
  unsigned int u = __float_as_uint(f);
  u += 0x7FFFu + ((u >> 16) & 1u);          // round-to-nearest-even
  return (unsigned short)(u >> 16);
}
__device__ __forceinline__ unsigned int pack2(float a, float b) {
  return (unsigned int)bfc(a) | ((unsigned int)bfc(b) << 16);
}
// monotonic float<->uint so native u32 atomic min/max order like floats
__device__ __forceinline__ unsigned int fenc(float f) {
  unsigned int u = __float_as_uint(f);
  return (u & 0x80000000u) ? ~u : (u | 0x80000000u);
}
__device__ __forceinline__ float fdec(unsigned int u) {
  u = (u & 0x80000000u) ? (u & 0x7FFFFFFFu) : ~u;
  return __uint_as_float(u);
}
__device__ __forceinline__ v8f wmma_bf16(v16bf a, v16bf b, v8f c) {
  return __builtin_amdgcn_wmma_f32_16x16x32_bf16(false, a, false, b, (short)0, c,
                                                 false, false);
}
// A fragment (16x32 bf16, MxK) from LDS row-major tile.
// lanes 0-15: M=lane, K=0..15; lanes 16-31: M=lane-16, K=16..31
__device__ __forceinline__ v16bf load_a_lds(const unsigned short* buf, int stride,
                                            int colbase, int lane) {
  int m = lane & 15;
  int koff = (lane & 16) ? 16 : 0;
  const unsigned short* row = buf + m * stride + colbase;
  uint4 lo = *(const uint4*)(row + koff);        // v0..3 : K = koff+0..7
  uint4 hi = *(const uint4*)(row + koff + 8);    // v4..7 : K = koff+8..15
  v8u t;
  t[0] = lo.x; t[1] = lo.y; t[2] = lo.z; t[3] = lo.w;
  t[4] = hi.x; t[5] = hi.y; t[6] = hi.z; t[7] = hi.w;
  return __builtin_bit_cast(v16bf, t);
}
// B fragment (32x16 bf16, KxN) from weight W[g][f] (row-major, fin inner):
// B[k][n] = W[g0+n][kbase+k]. lanes 0-15: K=0..15; lanes 16-31: K=16..31.
__device__ __forceinline__ v16bf load_b_glb(const unsigned short* W, int fin,
                                            int g0, int kbase, int lane) {
  int n = lane & 15;
  int koff = (lane & 16) ? 16 : 0;
  const unsigned short* p = W + (size_t)(g0 + n) * fin + kbase + koff;
  uint4 lo = *(const uint4*)(p);
  uint4 hi = *(const uint4*)(p + 8);
  v8u t;
  t[0] = lo.x; t[1] = lo.y; t[2] = lo.z; t[3] = lo.w;
  t[4] = hi.x; t[5] = hi.y; t[6] = hi.z; t[7] = hi.w;
  return __builtin_bit_cast(v16bf, t);
}

// ---------- small prep kernels ----------
__global__ void init_kernel(float* s, float* q, unsigned int* mn,
                            unsigned int* mx, unsigned int* cnt) {
  int i = blockIdx.x * blockDim.x + threadIdx.x;
  if (i < NN * HD) { s[i] = 0.f; q[i] = 0.f; mn[i] = 0xFFFFFFFFu; mx[i] = 0u; }
  if (i < NN) cnt[i] = 0u;
}

__global__ void cvt_kernel(const float* __restrict__ src,
                           unsigned short* __restrict__ dst, int n) {
  int i = blockIdx.x * blockDim.x + threadIdx.x;
  if (i < n) dst[i] = bfc(src[i]);
}

// e5[r][f] = emb[r] . edge_w[f] + edge_b[f]  (only 5 distinct bond types)
// avg_log from deg_hist
__global__ void prep_kernel(const float* __restrict__ emb,
                            const float* __restrict__ ew,
                            const float* __restrict__ eb,
                            const int* __restrict__ dh,
                            float* __restrict__ e5, float* __restrict__ avgp) {
  int tid = threadIdx.x;
  if (tid < 5 * FT) {
    int r = tid / FT, f = tid % FT;
    float acc = eb[f];
    for (int k = 0; k < HD; ++k) acc += emb[r * HD + k] * ew[f * HD + k];
    e5[r * FT + f] = acc;
  } else if (tid == 5 * FT) {
    float sl = 0.f, sc = 0.f;
    for (int i = 0; i < 17; ++i) {
      float d = (float)dh[i];
      sl += logf((float)i + 1.f) * d;
      sc += d;
    }
    avgp[0] = sl / sc;
  }
}

__global__ void deg_kernel(const int* __restrict__ dstI, unsigned int* cnt) {
  int i = blockIdx.x * blockDim.x + threadIdx.x;
  if (i < NE) atomicAdd(&cnt[dstI[i]], 1u);
}

// ---------- edge kernel: pre-MLP (WMMA) + scatter stats ----------
__global__ __launch_bounds__(128) void edge_kernel(
    const float* __restrict__ x, const int* __restrict__ bond,
    const int* __restrict__ srcI, const int* __restrict__ dstI,
    const float* __restrict__ e5,
    const unsigned short* __restrict__ w0b, const float* __restrict__ b0,
    const unsigned short* __restrict__ w1b, const float* __restrict__ b1,
    float* __restrict__ sumB, float* __restrict__ sqB,
    unsigned int* __restrict__ mnB, unsigned int* __restrict__ mxB) {
  const int RS = 296;  // 288 cols (+8 pad): [xdst 128 | xsrc 128 | e 32]
  __shared__ __align__(16) unsigned short stage[4][16][296];
  __shared__ __align__(16) unsigned short h1s[4][16][40];
  __shared__ int dnode[4][16];
  int wv = threadIdx.x >> 5;
  int lane = threadIdx.x & 31;
  int base = blockIdx.x * 64 + wv * 16;

  // gather + bf16 staging (16 edges per wave)
  for (int i = 0; i < 16; ++i) {
    int e = base + i;
    int nd = dstI[e];
    int ns = srcI[e];
    if (lane == 0) dnode[wv][i] = nd;
    float4 vd = *(const float4*)(x + (size_t)nd * HD + lane * 4);
    float4 vs = *(const float4*)(x + (size_t)ns * HD + lane * 4);
    uint2 pd; pd.x = pack2(vd.x, vd.y); pd.y = pack2(vd.z, vd.w);
    uint2 ps; ps.x = pack2(vs.x, vs.y); ps.y = pack2(vs.z, vs.w);
    *(uint2*)&stage[wv][i][lane * 4] = pd;
    *(uint2*)&stage[wv][i][128 + lane * 4] = ps;
    stage[wv][i][256 + lane] = bfc(e5[bond[e] * FT + lane]);
  }
  __syncthreads();

  for (int t = 0; t < NT; ++t) {
    v16bf a0 = load_a_lds(&stage[wv][0][0], RS, t * 32, lane);        // x_dst slice
    v16bf a1 = load_a_lds(&stage[wv][0][0], RS, 128 + t * 32, lane);  // x_src slice
    v16bf a2 = load_a_lds(&stage[wv][0][0], RS, 256, lane);           // edge attr
    const unsigned short* W0 = w0b + t * 32 * 96;
    // two independent accumulator chains (n0 = 0,1), shared A fragments
    v8f acc0 = {}, acc1 = {};
    acc0 = wmma_bf16(a0, load_b_glb(W0, 96, 0, 0, lane), acc0);
    acc1 = wmma_bf16(a0, load_b_glb(W0, 96, 16, 0, lane), acc1);
    acc0 = wmma_bf16(a1, load_b_glb(W0, 96, 0, 32, lane), acc0);
    acc1 = wmma_bf16(a1, load_b_glb(W0, 96, 16, 32, lane), acc1);
    acc0 = wmma_bf16(a2, load_b_glb(W0, 96, 0, 64, lane), acc0);
    acc1 = wmma_bf16(a2, load_b_glb(W0, 96, 16, 64, lane), acc1);
    {
      float bias0 = b0[t * 32 + (lane & 15)];
      float bias1 = b0[t * 32 + 16 + (lane & 15)];
      int n = lane & 15;
      int mh = (lane >> 4) << 3;
#pragma unroll
      for (int v = 0; v < 8; ++v) {
        h1s[wv][v + mh][n] = bfc(fmaxf(acc0[v] + bias0, 0.f));
        h1s[wv][v + mh][16 + n] = bfc(fmaxf(acc1[v] + bias1, 0.f));
      }
    }
    __syncthreads();
    v16bf ah = load_a_lds(&h1s[wv][0][0], 40, 0, lane);
    v8f o0 = {}, o1 = {};
    o0 = wmma_bf16(ah, load_b_glb(w1b + t * 32 * 32, 32, 0, 0, lane), o0);
    o1 = wmma_bf16(ah, load_b_glb(w1b + t * 32 * 32, 32, 16, 0, lane), o1);
    {
      float bias0 = b1[t * 32 + (lane & 15)];
      float bias1 = b1[t * 32 + 16 + (lane & 15)];
      int feat0 = t * 32 + (lane & 15);
      int mh = (lane >> 4) << 3;
#pragma unroll
      for (int v = 0; v < 8; ++v) {
        int node = dnode[wv][v + mh];
        size_t idxb = (size_t)node * HD + feat0;
        float val0 = o0[v] + bias0;
        float val1 = o1[v] + bias1;
        atomicAdd(&sumB[idxb], val0);
        atomicAdd(&sqB[idxb], val0 * val0);
        atomicMin(&mnB[idxb], fenc(val0));
        atomicMax(&mxB[idxb], fenc(val0));
        atomicAdd(&sumB[idxb + 16], val1);
        atomicAdd(&sqB[idxb + 16], val1 * val1);
        atomicMin(&mnB[idxb + 16], fenc(val1));
        atomicMax(&mxB[idxb + 16], fenc(val1));
      }
    }
    __syncthreads();
  }
}

// ---------- node kernel: scalers + post-MLP + linear + LN (WMMA) ----------
__global__ __launch_bounds__(64) void node_kernel(
    const float* __restrict__ x,
    const float* __restrict__ sumB, const float* __restrict__ sqB,
    const unsigned int* __restrict__ mnB, const unsigned int* __restrict__ mxB,
    const unsigned int* __restrict__ cnt, const float* __restrict__ avgp,
    const unsigned short* __restrict__ w0b, const float* __restrict__ b0,
    const unsigned short* __restrict__ w1b, const float* __restrict__ b1,
    const unsigned short* __restrict__ lwb, const float* __restrict__ lb,
    const float* __restrict__ lng, const float* __restrict__ lnbv,
    float* __restrict__ out) {
  const int AS = 520;  // 512 post-in cols (+8 pad)
  __shared__ __align__(16) unsigned short astage[2][16][520];
  __shared__ __align__(16) unsigned short h1s[2][16][40];
  __shared__ __align__(16) unsigned short pls[2][16][136];  // pre-linear 128 (+8)
  __shared__ float scal[2][16][4];
  int wv = threadIdx.x >> 5;
  int lane = threadIdx.x & 31;
  int base = blockIdx.x * 32 + wv * 16;
  float avg_log = avgp[0];
  if (lane < 16) {
    unsigned int c = cnt[base + lane];
    float cf = (float)c;
    float c1 = fmaxf(cf, 1.f);
    float logd = logf(c1 + 1.f);
    scal[wv][lane][0] = logd / avg_log;
    scal[wv][lane][1] = avg_log / logd;
    scal[wv][lane][2] = c1;
    scal[wv][lane][3] = (c > 0u) ? 1.f : 0.f;
  }
  __syncthreads();

  for (int t = 0; t < NT; ++t) {
    // build 512-wide PNA rows: [x | s mean mn mx std | agg*s1 | agg*s2]
    for (int j = 0; j < 16; ++j) {
      int r = lane & 15;
      int f = j * 2 + (lane >> 4);
      size_t idx = (size_t)(base + r) * HD + t * 32 + f;
      float c1 = scal[wv][r][2];
      float has = scal[wv][r][3];
      float s1 = scal[wv][r][0];
      float s2 = scal[wv][r][1];
      float s = sumB[idx];
      float mean = s / c1;
      float m2 = sqB[idx] / c1;
      float sd = sqrtf(fmaxf(m2 - mean * mean, 0.f) + 1e-5f);
      float mn = (has > 0.5f) ? fdec(mnB[idx]) : 0.f;
      float mx = (has > 0.5f) ? fdec(mxB[idx]) : 0.f;
      unsigned short* row = &astage[wv][r][0];
      row[f] = bfc(x[idx]);
      float ag[5] = {s, mean, mn, mx, sd};
#pragma unroll
      for (int a = 0; a < 5; ++a) {
        row[32 + a * 32 + f] = bfc(ag[a]);
        row[192 + a * 32 + f] = bfc(ag[a] * s1);
        row[352 + a * 32 + f] = bfc(ag[a] * s2);
      }
    }
    __syncthreads();
    const unsigned short* W0 = w0b + t * 32 * 512;
    // dual accumulators over one K loop: A fragment loaded once per kb,
    // two independent WMMA chains to hide matrix-pipe latency
    v8f acc0 = {}, acc1 = {};
    for (int kb = 0; kb < 16; ++kb) {
      v16bf a = load_a_lds(&astage[wv][0][0], AS, kb * 32, lane);
      acc0 = wmma_bf16(a, load_b_glb(W0, 512, 0, kb * 32, lane), acc0);
      acc1 = wmma_bf16(a, load_b_glb(W0, 512, 16, kb * 32, lane), acc1);
    }
    {
      float bias0 = b0[t * 32 + (lane & 15)];
      float bias1 = b0[t * 32 + 16 + (lane & 15)];
      int n = lane & 15;
      int mh = (lane >> 4) << 3;
#pragma unroll
      for (int v = 0; v < 8; ++v) {
        h1s[wv][v + mh][n] = bfc(fmaxf(acc0[v] + bias0, 0.f));
        h1s[wv][v + mh][16 + n] = bfc(fmaxf(acc1[v] + bias1, 0.f));
      }
    }
    __syncthreads();
    v16bf ah = load_a_lds(&h1s[wv][0][0], 40, 0, lane);
    v8f o0 = {}, o1 = {};
    o0 = wmma_bf16(ah, load_b_glb(w1b + t * 32 * 32, 32, 0, 0, lane), o0);
    o1 = wmma_bf16(ah, load_b_glb(w1b + t * 32 * 32, 32, 16, 0, lane), o1);
    {
      float bias0 = b1[t * 32 + (lane & 15)];
      float bias1 = b1[t * 32 + 16 + (lane & 15)];
      int col0 = t * 32 + (lane & 15);
      int mh = (lane >> 4) << 3;
#pragma unroll
      for (int v = 0; v < 8; ++v) {
        pls[wv][v + mh][col0] = bfc(o0[v] + bias0);
        pls[wv][v + mh][col0 + 16] = bfc(o1[v] + bias1);
      }
    }
    __syncthreads();
  }

  // final 128x128 linear, then LayerNorm + ReLU + residual
  v16bf afr0 = load_a_lds(&pls[wv][0][0], 136, 0, lane);
  v16bf afr1 = load_a_lds(&pls[wv][0][0], 136, 32, lane);
  v16bf afr2 = load_a_lds(&pls[wv][0][0], 136, 64, lane);
  v16bf afr3 = load_a_lds(&pls[wv][0][0], 136, 96, lane);
  float dacc[8][8];
  float rs[8], rq[8];
#pragma unroll
  for (int v = 0; v < 8; ++v) { rs[v] = 0.f; rq[v] = 0.f; }
#pragma unroll
  for (int nt = 0; nt < 8; ++nt) {
    v8f acc = {};
    acc = wmma_bf16(afr0, load_b_glb(lwb, 128, nt * 16, 0, lane), acc);
    acc = wmma_bf16(afr1, load_b_glb(lwb, 128, nt * 16, 32, lane), acc);
    acc = wmma_bf16(afr2, load_b_glb(lwb, 128, nt * 16, 64, lane), acc);
    acc = wmma_bf16(afr3, load_b_glb(lwb, 128, nt * 16, 96, lane), acc);
    float bias = lb[nt * 16 + (lane & 15)];
#pragma unroll
    for (int v = 0; v < 8; ++v) {
      float val = acc[v] + bias;
      dacc[nt][v] = val;
      rs[v] += val;
      rq[v] += val * val;
    }
  }
  // row reduction: lanes of same 16-lane half share row M = v (+8 for hi half)
#pragma unroll
  for (int v = 0; v < 8; ++v) {
    float s = rs[v], q = rq[v];
#pragma unroll
    for (int m = 8; m >= 1; m >>= 1) {
      s += __shfl_xor(s, m, 32);
      q += __shfl_xor(q, m, 32);
    }
    float mu = s * (1.f / 128.f);
    float var = q * (1.f / 128.f) - mu * mu;
    rs[v] = mu;
    rq[v] = rsqrtf(var + 1e-5f);
  }
  int mh = (lane >> 4) << 3;
#pragma unroll
  for (int nt = 0; nt < 8; ++nt) {
    int col = nt * 16 + (lane & 15);
    float g = lng[col], bb = lnbv[col];
#pragma unroll
    for (int v = 0; v < 8; ++v) {
      int node = base + v + mh;
      float o = (dacc[nt][v] - rs[v]) * rq[v] * g + bb;
      out[(size_t)node * HD + col] = x[(size_t)node * HD + col] + fmaxf(o, 0.f);
    }
  }
}

// ---------- launch ----------
extern "C" void kernel_launch(void* const* d_in, const int* in_sizes, int n_in,
                              void* d_out, int out_size, void* d_ws,
                              size_t ws_size, hipStream_t stream) {
  (void)in_sizes; (void)n_in; (void)out_size; (void)ws_size;
  const float* atom_x  = (const float*)d_in[0];
  const int*   bond_x  = (const int*)d_in[1];
  const int*   eidx    = (const int*)d_in[2];
  const int*   deg     = (const int*)d_in[3];
  const float* emb     = (const float*)d_in[4];
  const float* edge_w  = (const float*)d_in[5];
  const float* edge_b  = (const float*)d_in[6];
  const float* pre_w0  = (const float*)d_in[7];
  const float* pre_b0  = (const float*)d_in[8];
  const float* pre_w1  = (const float*)d_in[9];
  const float* pre_b1  = (const float*)d_in[10];
  const float* post_w0 = (const float*)d_in[11];
  const float* post_b0 = (const float*)d_in[12];
  const float* post_w1 = (const float*)d_in[13];
  const float* post_b1 = (const float*)d_in[14];
  const float* lin_w   = (const float*)d_in[15];
  const float* lin_b   = (const float*)d_in[16];
  const float* ln_g    = (const float*)d_in[17];
  const float* ln_b    = (const float*)d_in[18];
  const int* srcI = eidx;
  const int* dstI = eidx + NE;

  char* ws = (char*)d_ws;
  const size_t SZ = (size_t)NN * HD * sizeof(float);
  size_t off = 0;
  float* sumB = (float*)(ws + off); off += SZ;
  float* sqB  = (float*)(ws + off); off += SZ;
  unsigned int* mnB = (unsigned int*)(ws + off); off += SZ;
  unsigned int* mxB = (unsigned int*)(ws + off); off += SZ;
  unsigned int* cnt = (unsigned int*)(ws + off); off += (size_t)NN * 4;
  float* e5   = (float*)(ws + off); off += 1024;
  float* avgp = (float*)(ws + off); off += 256;
  unsigned short* w0e = (unsigned short*)(ws + off); off += (size_t)2 * 4 * 32 * 96;
  unsigned short* w1e = (unsigned short*)(ws + off); off += (size_t)2 * 4 * 32 * 32;
  unsigned short* w0p = (unsigned short*)(ws + off); off += (size_t)2 * 4 * 32 * 512;
  unsigned short* w1p = (unsigned short*)(ws + off); off += (size_t)2 * 4 * 32 * 32;
  unsigned short* wl  = (unsigned short*)(ws + off); off += (size_t)2 * 128 * 128;

  init_kernel<<<(NN * HD + 255) / 256, 256, 0, stream>>>(sumB, sqB, mnB, mxB, cnt);
  cvt_kernel<<<(4 * 32 * 96 + 255) / 256, 256, 0, stream>>>(pre_w0, w0e, 4 * 32 * 96);
  cvt_kernel<<<(4 * 32 * 32 + 255) / 256, 256, 0, stream>>>(pre_w1, w1e, 4 * 32 * 32);
  cvt_kernel<<<(4 * 32 * 512 + 255) / 256, 256, 0, stream>>>(post_w0, w0p, 4 * 32 * 512);
  cvt_kernel<<<(4 * 32 * 32 + 255) / 256, 256, 0, stream>>>(post_w1, w1p, 4 * 32 * 32);
  cvt_kernel<<<(128 * 128 + 255) / 256, 256, 0, stream>>>(lin_w, wl, 128 * 128);
  prep_kernel<<<1, 192, 0, stream>>>(emb, edge_w, edge_b, deg, e5, avgp);
  deg_kernel<<<(NE + 255) / 256, 256, 0, stream>>>(dstI, cnt);
  edge_kernel<<<NE / 64, 128, 0, stream>>>(atom_x, bond_x, srcI, dstI, e5, w0e,
                                           pre_b0, w1e, pre_b1, sumB, sqB, mnB, mxB);
  node_kernel<<<NN / 32, 64, 0, stream>>>(atom_x, sumB, sqB, mnB, mxB, cnt, avgp,
                                          w0p, post_b0, w1p, post_b1, wl, lin_b,
                                          ln_g, ln_b, (float*)d_out);
}
